// MultiSimilarityLoss_6622839570702
// MI455X (gfx1250) — compile-verified, matrix operand order checked by reading, not run on hardware
//
#include <hip/hip_runtime.h>
#include <hip/hip_bf16.h>
#include <math.h>

#define MS_N 8192
#define MS_D 512
#define MS_C 128
#define MS_ALPHA 2.0f
#define MS_BETA  50.0f
#define MS_BASE  0.5f
#define MS_EPS   0.1f

typedef __attribute__((ext_vector_type(16))) _Float16 v16h;
typedef __attribute__((ext_vector_type(8)))  _Float16 v8h;
typedef __attribute__((ext_vector_type(8)))  float    v8f;

// Monotone float <-> uint key (enables atomicMax/Min on shared uints for floats).
__device__ __forceinline__ unsigned ms_fkey(float f) {
    unsigned u = __float_as_uint(f);
    return (u & 0x80000000u) ? ~u : (u | 0x80000000u);
}
__device__ __forceinline__ float ms_fdecode(unsigned k) {
    unsigned u = (k & 0x80000000u) ? (k ^ 0x80000000u) : ~k;
    return __uint_as_float(u);
}

// ---------------------------------------------------------------------------
// Kernel 1: L2-normalize rows of features (fp32) -> fp16 matrix Fh [N, D].
// One wave per row; 16 contiguous floats per lane; shuffle reduction.
// ---------------------------------------------------------------------------
__global__ __launch_bounds__(256)
void ms_normalize_kernel(const float* __restrict__ f, _Float16* __restrict__ fh) {
    const int row  = blockIdx.x * 8 + (threadIdx.x >> 5);
    const int lane = threadIdx.x & 31;
    const float* src = f + (size_t)row * MS_D + lane * 16;
    float vals[16];
#pragma unroll
    for (int q = 0; q < 4; ++q) {
        float4 t = ((const float4*)src)[q];
        vals[4*q+0] = t.x; vals[4*q+1] = t.y; vals[4*q+2] = t.z; vals[4*q+3] = t.w;
    }
    float ss = 0.f;
#pragma unroll
    for (int q = 0; q < 16; ++q) ss += vals[q] * vals[q];
#pragma unroll
    for (int d = 1; d < 32; d <<= 1) ss += __shfl_xor(ss, d, 32);
    const float inv = rsqrtf(ss);
    v8h h0, h1;
#pragma unroll
    for (int q = 0; q < 8; ++q) {
        h0[q] = (_Float16)(vals[q]     * inv);
        h1[q] = (_Float16)(vals[8 + q] * inv);
    }
    v8h* dst = (v8h*)(fh + (size_t)row * MS_D + lane * 16);
    dst[0] = h0;
    dst[1] = h1;
}

// ---------------------------------------------------------------------------
// Tiny init + label histogram (for pos_mask.any / neg_mask.any validity).
// ---------------------------------------------------------------------------
__global__ void ms_init_kernel(int* counts, float* gsum, int* gcnt) {
    const int t = threadIdx.x;
    if (t < MS_C)      counts[t] = 0;
    if (t == MS_C)     *gsum = 0.f;
    if (t == MS_C + 1) *gcnt = 0;
}

__global__ void ms_hist_kernel(const int* __restrict__ labels, int* __restrict__ counts) {
    const int i = blockIdx.x * 256 + threadIdx.x;
    if (i < MS_N) atomicAdd(&counts[labels[i]], 1);
}

// ---------------------------------------------------------------------------
// Fused WMMA GEMM + masked row-reduction.
// Block owns 32 rows (A panel in LDS); 8 waves stripe the 512 column tiles in
// pairs. Each wave: 2 M-tiles x 2 N-tiles -> 4 v_wmma_f32_16x16x32_f16 per
// K-step, sharing both A (ds_load_b128 x4) and B (global_load_b128 x4; the
// second N-tile is a fixed +16KB offset of the same pointer).
//
//   MODE 0: max_neg[i]  = max over j (label differs) of sim
//   MODE 1: rowstat=max_neg -> min_pos[i] and pos_sum[i] over pos_keep
//   MODE 2: rowstat=min_pos -> neg_sum[i] over neg_keep
//
// Row reductions are carried in per-lane registers across the whole N sweep
// (each lane owns a fixed column subset); the cross-lane shuffle reduction +
// LDS atomic combine happens once per kernel at the end.
//
// Per-lane operand layout (CDNA5 ISA 7.12.2, 16-bit A 16x32):
//   lanes 0-15 : M=lane,    K in {k0..k0+7, k0+16..k0+23}
//   lanes 16-31: M=lane-16, K in {k0+8..k0+15, k0+24..k0+31}
// B (32x16 KxN) mirrored with lane%16 = N. Since sim = Fh * Fh^T, A and B use
// the same per-lane fetch pattern. C/D: lane%16 = N, M = r + 8*(lane/16).
// ---------------------------------------------------------------------------
template <int MODE>
__global__ __launch_bounds__(256)
void ms_gemm_kernel(const _Float16* __restrict__ Fh,
                    const int*      __restrict__ labels,
                    const float*    __restrict__ rowstat,
                    float*          __restrict__ out_a,
                    float*          __restrict__ out_b) {
    __shared__ _Float16 sA[32 * MS_D];   // 32 KB A panel
    __shared__ int      sLbl[32];
    __shared__ unsigned uRed[32];        // keyed min/max accumulator
    __shared__ float    fRed[32];        // float sum accumulator
    __shared__ float    sStat[32];       // per-row threshold (modes 1,2)

    const int tid = threadIdx.x;
    const int m0  = blockIdx.x * 32;

    // Cooperative copy of the 32x512 f16 A panel into LDS (b128 transfers).
    {
        const v8h* g = (const v8h*)(Fh + (size_t)m0 * MS_D);
        v8h*       s = (v8h*)sA;
#pragma unroll
        for (int it = 0; it < (32 * MS_D / 8) / 256; ++it)
            s[tid + it * 256] = g[tid + it * 256];
    }
    if (tid < 32) {
        sLbl[tid] = labels[m0 + tid];
        if (MODE == 0) { uRed[tid] = ms_fkey(-1e30f); }
        if (MODE == 1) { uRed[tid] = ms_fkey( 1e30f); fRed[tid] = 0.f; sStat[tid] = rowstat[m0 + tid]; }
        if (MODE == 2) { fRed[tid] = 0.f; sStat[tid] = rowstat[m0 + tid]; }
    }
    __syncthreads();

    const int lane = tid & 31;
    const int wv   = tid >> 5;
    const int half = lane >> 4;
    const int lm   = lane & 15;
    const int koff = half * 8;   // K-chunk select per the 16-bit operand layout

    // Preload this lane's row metadata (invariant across the N sweep).
    int   rlbl[2][8];
    float rstat[2][8];
#pragma unroll
    for (int t = 0; t < 2; ++t)
#pragma unroll
        for (int r = 0; r < 8; ++r) {
            const int mloc = 16 * t + 8 * half + r;
            rlbl[t][r] = sLbl[mloc];
            if (MODE != 0) rstat[t][r] = sStat[mloc];
        }

    // Per-lane running reductions over this lane's columns.
    float run0[2][8];   // MODE 0: max      | MODE 1: min
    float run1[2][8];   // MODE 1/2: sum
#pragma unroll
    for (int t = 0; t < 2; ++t)
#pragma unroll
        for (int r = 0; r < 8; ++r) {
            run0[t][r] = (MODE == 1) ? 1e30f : -1e30f;
            run1[t][r] = 0.f;
        }

    for (int nt = 2 * wv; nt < MS_N / 16; nt += 16) {
        const int j0  = nt * 16 + lm;        // lane's column in N-tile 0
        const int j1  = j0 + 16;             // lane's column in N-tile 1
        const int cl0 = labels[j0];
        const int cl1 = labels[j1];
        const _Float16* bp = Fh + (size_t)j0 * MS_D + koff;

        v8f acc[2][2] = {};                  // [n][t]
#pragma unroll 2
        for (int k0 = 0; k0 < MS_D; k0 += 32) {
            v8h b0lo = *(const v8h*)(bp + k0);
            v8h b0hi = *(const v8h*)(bp + k0 + 16);
            v8h b1lo = *(const v8h*)(bp + 16 * MS_D + k0);
            v8h b1hi = *(const v8h*)(bp + 16 * MS_D + k0 + 16);
            v16h B0, B1;
#pragma unroll
            for (int q = 0; q < 8; ++q) {
                B0[q] = b0lo[q]; B0[q + 8] = b0hi[q];
                B1[q] = b1lo[q]; B1[q + 8] = b1hi[q];
            }

            const _Float16* ap = sA + (size_t)lm * MS_D + k0 + koff;
            v8h a0lo = *(const v8h*)(ap);
            v8h a0hi = *(const v8h*)(ap + 16);
            v8h a1lo = *(const v8h*)(ap + 16 * MS_D);
            v8h a1hi = *(const v8h*)(ap + 16 * MS_D + 16);
            v16h A0, A1;
#pragma unroll
            for (int q = 0; q < 8; ++q) {
                A0[q] = a0lo[q]; A0[q + 8] = a0hi[q];
                A1[q] = a1lo[q]; A1[q + 8] = a1hi[q];
            }

            acc[0][0] = __builtin_amdgcn_wmma_f32_16x16x32_f16(
                false, A0, false, B0, (short)0, acc[0][0], false, false);
            acc[0][1] = __builtin_amdgcn_wmma_f32_16x16x32_f16(
                false, A1, false, B0, (short)0, acc[0][1], false, false);
            acc[1][0] = __builtin_amdgcn_wmma_f32_16x16x32_f16(
                false, A0, false, B1, (short)0, acc[1][0], false, false);
            acc[1][1] = __builtin_amdgcn_wmma_f32_16x16x32_f16(
                false, A1, false, B1, (short)0, acc[1][1], false, false);
        }

        // Register-only epilogue: fold this 32x32 sim block into the running
        // per-row reductions. jlocal vs (16t+r) implements the j != i test.
#pragma unroll
        for (int n = 0; n < 2; ++n) {
            const int cl     = n ? cl1 : cl0;
            const int jj     = n ? j1 : j0;
            const int jlocal = jj - m0 - 8 * half;
#pragma unroll
            for (int t = 0; t < 2; ++t)
#pragma unroll
                for (int r = 0; r < 8; ++r) {
                    const float s = acc[n][t][r];
                    if (MODE == 0) {
                        run0[t][r] = fmaxf(run0[t][r],
                                           (rlbl[t][r] != cl) ? s : -1e30f);
                    } else if (MODE == 1) {
                        const bool keep = (rlbl[t][r] == cl) &&
                                          (jlocal != 16 * t + r) &&
                                          (s < rstat[t][r] + MS_EPS);
                        run0[t][r] = fminf(run0[t][r], keep ? s : 1e30f);
                        run1[t][r] += keep ? __expf(-MS_ALPHA * (s - MS_BASE)) : 0.f;
                    } else {
                        const bool keep = (rlbl[t][r] != cl) &&
                                          (s > rstat[t][r] - MS_EPS);
                        run1[t][r] += keep ? __expf(MS_BETA * (s - MS_BASE)) : 0.f;
                    }
                }
        }
    }

    // One-time cross-lane reduction (within each 16-lane half) + LDS combine.
#pragma unroll
    for (int t = 0; t < 2; ++t)
#pragma unroll
        for (int r = 0; r < 8; ++r) {
            const int mloc = 16 * t + 8 * half + r;
            if (MODE == 0) {
                float v = run0[t][r];
#pragma unroll
                for (int d = 1; d < 16; d <<= 1) v = fmaxf(v, __shfl_xor(v, d, 32));
                if (lm == 0) atomicMax(&uRed[mloc], ms_fkey(v));
            } else if (MODE == 1) {
                float v = run0[t][r];
                float e = run1[t][r];
#pragma unroll
                for (int d = 1; d < 16; d <<= 1) {
                    v = fminf(v, __shfl_xor(v, d, 32));
                    e += __shfl_xor(e, d, 32);
                }
                if (lm == 0) { atomicMin(&uRed[mloc], ms_fkey(v)); atomicAdd(&fRed[mloc], e); }
            } else {
                float e = run1[t][r];
#pragma unroll
                for (int d = 1; d < 16; d <<= 1) e += __shfl_xor(e, d, 32);
                if (lm == 0) atomicAdd(&fRed[mloc], e);
            }
        }
    __syncthreads();

    if (tid < 32) {
        if (MODE == 0) { out_a[m0 + tid] = ms_fdecode(uRed[tid]); }
        if (MODE == 1) { out_a[m0 + tid] = ms_fdecode(uRed[tid]); out_b[m0 + tid] = fRed[tid]; }
        if (MODE == 2) { out_b[m0 + tid] = fRed[tid]; }
    }
}

// ---------------------------------------------------------------------------
// Row loss + validity, accumulated into global scalars.
// valid = (count>1) & (count<N) & (pos_sum>0) & (neg_sum>0)
// ---------------------------------------------------------------------------
__global__ __launch_bounds__(256)
void ms_rowloss_kernel(const int*   __restrict__ labels,
                       const int*   __restrict__ counts,
                       const float* __restrict__ possum,
                       const float* __restrict__ negsum,
                       float* __restrict__ gsum, int* __restrict__ gcnt) {
    const int i = blockIdx.x * 256 + threadIdx.x;
    float loss = 0.f;
    int   v    = 0;
    if (i < MS_N) {
        const int   c  = counts[labels[i]];
        const float ps = possum[i];
        const float ns = negsum[i];
        const bool valid = (c > 1) && (c < MS_N) && (ps > 0.f) && (ns > 0.f);
        if (valid) {
            loss = log1pf(ps) / MS_ALPHA + log1pf(ns) / MS_BETA;
            v = 1;
        }
    }
#pragma unroll
    for (int d = 1; d < 32; d <<= 1) { loss += __shfl_xor(loss, d, 32); v += __shfl_xor(v, d, 32); }
    __shared__ float ls[8];
    __shared__ int   vs[8];
    const int wv = threadIdx.x >> 5;
    if ((threadIdx.x & 31) == 0) { ls[wv] = loss; vs[wv] = v; }
    __syncthreads();
    if (threadIdx.x == 0) {
        float L = 0.f; int V = 0;
#pragma unroll
        for (int k = 0; k < 8; ++k) { L += ls[k]; V += vs[k]; }
        atomicAdd(gsum, L);
        atomicAdd(gcnt, V);
    }
}

__global__ void ms_final_kernel(const float* gsum, const int* gcnt, float* out) {
    if (threadIdx.x == 0 && blockIdx.x == 0) {
        int c = *gcnt;
        if (c < 1) c = 1;
        out[0] = *gsum / (float)c;
    }
}

// ---------------------------------------------------------------------------
// Launch. Workspace layout (~8.5 MB):
//   Fh[N*D] f16 | max_neg[N] | min_pos[N] | pos_sum[N] | neg_sum[N] |
//   counts[C] | gsum | gcnt
// ---------------------------------------------------------------------------
extern "C" void kernel_launch(void* const* d_in, const int* in_sizes, int n_in,
                              void* d_out, int out_size, void* d_ws, size_t ws_size,
                              hipStream_t stream) {
    (void)in_sizes; (void)n_in; (void)out_size; (void)ws_size;
    const float* features = (const float*)d_in[0];
    const int*   labels   = (const int*)d_in[1];
    float*       out      = (float*)d_out;

    char* ws = (char*)d_ws;
    _Float16* Fh = (_Float16*)ws;
    size_t off = (size_t)MS_N * MS_D * sizeof(_Float16);
    float* maxneg = (float*)(ws + off); off += (size_t)MS_N * 4;
    float* minpos = (float*)(ws + off); off += (size_t)MS_N * 4;
    float* possum = (float*)(ws + off); off += (size_t)MS_N * 4;
    float* negsum = (float*)(ws + off); off += (size_t)MS_N * 4;
    int*   counts = (int*)(ws + off);   off += (size_t)MS_C * 4;
    float* gsum   = (float*)(ws + off); off += 4;
    int*   gcnt   = (int*)(ws + off);   off += 4;

    ms_init_kernel<<<1, 256, 0, stream>>>(counts, gsum, gcnt);
    ms_normalize_kernel<<<MS_N / 8, 256, 0, stream>>>(features, Fh);
    ms_hist_kernel<<<MS_N / 256, 256, 0, stream>>>(labels, counts);
    ms_gemm_kernel<0><<<MS_N / 32, 256, 0, stream>>>(Fh, labels, nullptr, maxneg, nullptr);
    ms_gemm_kernel<1><<<MS_N / 32, 256, 0, stream>>>(Fh, labels, maxneg, minpos, possum);
    ms_gemm_kernel<2><<<MS_N / 32, 256, 0, stream>>>(Fh, labels, minpos, nullptr, negsum);
    ms_rowloss_kernel<<<MS_N / 256, 256, 0, stream>>>(labels, counts, possum, negsum, gsum, gcnt);
    ms_final_kernel<<<1, 32, 0, stream>>>(gsum, gcnt, out);
}
